// AliasFreeConv_1331439862242
// MI455X (gfx1250) — matrix operand order repaired and checked by reading, not compiled, not run on hardware
//
#include <hip/hip_runtime.h>

typedef _Float16 v16h __attribute__((ext_vector_type(16)));
typedef float    v8f  __attribute__((ext_vector_type(8)));

#define B_    8
#define CIN   512
#define COUT  512
#define H_    64
#define W_    64
#define OH    62
#define OW    62
#define KTOT  4608          // CIN * 3 * 3
#define NKS   144           // KTOT / 32
#define STYLE 512

// ---------------------------------------------------------------------------
// Kernel 1: s[b][ci] = style[b] . mod_w[ci] * (1/sqrt(512)) + mod_b[ci]
// ---------------------------------------------------------------------------
__global__ __launch_bounds__(256) void style_kernel(
    const float* __restrict__ style, const float* __restrict__ mod_w,
    const float* __restrict__ mod_b, float* __restrict__ S) {
  __shared__ float st[STYLE];
  const int b = blockIdx.x, tid = threadIdx.x;
  st[tid]       = style[b * STYLE + tid];
  st[tid + 256] = style[b * STYLE + tid + 256];
  __syncthreads();
  const float lin = 0.04419417382415922f;  // 1/sqrt(512)
  for (int ci = tid; ci < CIN; ci += 256) {
    const float* mw = mod_w + (size_t)ci * STYLE;
    float acc = 0.f;
    for (int j = 0; j < STYLE; ++j) acc += st[j] * mw[j];
    S[b * CIN + ci] = acc * lin + mod_b[ci];
  }
}

// ---------------------------------------------------------------------------
// Kernel 2: modulate + demodulate, pack f16 weights PRE-SWIZZLED into the
// A-fragment layout (contiguous kt per lane half):  Wsw[b][co>>4][ks][lane][idx], 16 halfs/lane
//   K order  : k = kk*512 + ci  (tap-outer -> implicit im2col slices)
//   lane     : (co&15) + 16*(kt>>4)
//   idx      : kt&15
// ---------------------------------------------------------------------------
__global__ __launch_bounds__(256) void wpack_kernel(
    const float* __restrict__ conv_w, const float* __restrict__ S,
    _Float16* __restrict__ Wsw) {
  const int bc = blockIdx.x;  // b*512 + co
  const int b = bc >> 9, co = bc & 511;
  const int tid = threadIdx.x;
  __shared__ float sS[CIN];
  __shared__ float red[256];
  sS[tid]       = S[b * CIN + tid];
  sS[tid + 256] = S[b * CIN + tid + 256];
  __syncthreads();
  const float wscale = 0.014731391274719742f;  // 1/sqrt(4608)
  const float* cw = conv_w + (size_t)co * KTOT;
  float v[18];
  float acc = 0.f;
#pragma unroll
  for (int i = 0; i < 18; ++i) {  // flat e = ci*9 + kk
    const int e = i * 256 + tid;
    const int ci = e / 9;
    const float w = wscale * cw[e] * sS[ci];
    v[i] = w;
    acc += w * w;
  }
  red[tid] = acc;
  __syncthreads();
  for (int s = 128; s > 0; s >>= 1) {
    if (tid < s) red[tid] += red[tid + s];
    __syncthreads();
  }
  const float scale = (1.f / sqrtf(red[0] + 1e-8f)) * (1.f / (1.f + 1e-8f));
  const int co_g = co >> 4, m = co & 15;
  _Float16* wb = Wsw + (size_t)(b * 32 + co_g) * (NKS * 32 * 16);
#pragma unroll
  for (int i = 0; i < 18; ++i) {
    const int e  = i * 256 + tid;
    const int ci = e / 9;
    const int kk = e - ci * 9;
    const int k  = kk * CIN + ci;
    const int ks = k >> 5, kt = k & 31;
    const int lane  = m + 16 * (kt >> 4);
    const int idx16 = kt & 15;
    wb[((size_t)ks * 32 + lane) * 16 + idx16] = (_Float16)(v[i] * scale);
  }
}

// ---------------------------------------------------------------------------
// Kernel 3: NCHW f32 -> NHWC f16 transpose of x (LDS tiled, both sides
// coalesced).  Xh[b][ih][iw][ci]
// ---------------------------------------------------------------------------
__global__ __launch_bounds__(256) void xcvt_kernel(
    const float* __restrict__ x, _Float16* __restrict__ Xh) {
  const int bih = blockIdx.x;  // b*64 + ih
  const int b = bih >> 6, ih = bih & 63;
  const int tid = threadIdx.x;
  __shared__ _Float16 sh[64 * 520];  // [iw][ci], pad 512->520
  for (int idx = tid; idx < CIN * 64; idx += 256) {
    const int ci = idx >> 6, iw = idx & 63;  // coalesced over iw
    sh[iw * 520 + ci] = (_Float16)x[((size_t)(b * CIN + ci) * H_ + ih) * W_ + iw];
  }
  __syncthreads();
  _Float16* dst = Xh + (size_t)bih * 64 * CIN;
  for (int idx = tid; idx < 64 * CIN; idx += 256) {
    const int iw = idx >> 9, ci = idx & 511;  // coalesced over ci
    dst[iw * CIN + ci] = sh[iw * 520 + ci];
  }
}

// ---------------------------------------------------------------------------
// Kernel 4: implicit-GEMM 3x3 conv, v_wmma_f32_16x16x32_f16.
// grid = (oh=62, Mtiles=4, batch=8); 256 thr = 8 wave32.
// Block tile 128(co) x 64(ow).  Wave tile 16x64: 1 A frag + 4 B frags = 4 WMMA.
//  - A frags: direct from pre-swizzled global (32 B contiguous / lane)
//  - B tile : async DMA global->LDS (double-buffered), stored transposed
//             [n][kt] so each B fragment is 32 contiguous bytes per lane.
// ---------------------------------------------------------------------------
__global__ __launch_bounds__(256) void conv_wmma_kernel(
    const _Float16* __restrict__ Xh, const _Float16* __restrict__ Wsw,
    const float* __restrict__ act_b, float* __restrict__ Cv) {
  const int oh = blockIdx.x;
  const int mt = blockIdx.y;
  const int b  = blockIdx.z;
  const int tid  = threadIdx.x;
  const int wave = tid >> 5;
  const int lane = tid & 31;

  __shared__ __attribute__((aligned(64))) _Float16 Bs[2][64 * 32];  // 2 x 4 KB

  v8f c0 = {}, c1 = {}, c2 = {}, c3 = {};

  // B staging: thread -> (row n = tid>>2, 16-byte chunk = tid&3)
  const int sn = tid >> 2;
  const int sc = tid & 3;
  const uint32_t lds0 =
      (uint32_t)(uintptr_t)(&Bs[0][0]) + (uint32_t)(sn * 64 + sc * 16);
  const _Float16* xrow = Xh + (size_t)(b * 64 + oh) * (64 * CIN) + sc * 8;

  // B fragment gather: lane -> col (lane&15), K-half (lane>>4)*16 (contig 32B)
  const int fnl = lane & 15;
  const int fkh = lane >> 4;

  // A fragment stream (pre-swizzled): 32 B per lane per K-slice
  const _Float16* aptr =
      Wsw + (((size_t)(b * 32 + mt * 8 + wave) * NKS) * 32 + lane) * 16;

  // ---- prologue: stage K-slice 0 (kk=0 -> kh=0,kw=0, ci0=0) into buf 0 ----
  {
    const uint64_t ga = (uint64_t)(uintptr_t)(xrow + (size_t)sn * CIN);
    asm volatile("global_load_async_to_lds_b128 %0, %1, off"
                 :: "v"(lds0), "v"(ga) : "memory");
  }
  asm volatile("s_wait_asynccnt 0x0" ::: "memory");
  __syncthreads();

  for (int ks = 0; ks < NKS; ++ks) {
    const int p = ks & 1;

    // issue async stage of next K-slice into the other buffer (overlapped)
    if (ks + 1 < NKS) {
      const int kkn = (ks + 1) >> 4;       // tap index 0..8
      const int ci0 = ((ks + 1) & 15) * 32;
      const int khn = kkn / 3;
      const int kwn = kkn - khn * 3;
      const uint64_t ga = (uint64_t)(uintptr_t)(
          xrow + ((size_t)(khn * 64) + sn + kwn) * CIN + ci0);
      const uint32_t ld = lds0 + (uint32_t)(((ks + 1) & 1) * 4096);
      asm volatile("global_load_async_to_lds_b128 %0, %1, off"
                   :: "v"(ld), "v"(ga) : "memory");
    }

    // A fragment: one 32-byte vector load from global (L2-resident)
    v16h a = *(const v16h*)(aptr + (size_t)ks * (32 * 16));

    // B fragments: 32 contiguous bytes per lane from transposed LDS tile
    const _Float16* bsrc = &Bs[p][0];
    v16h b0 = *(const v16h*)(bsrc + ((0 * 16 + fnl) * 32 + fkh * 16));
    v16h b1 = *(const v16h*)(bsrc + ((1 * 16 + fnl) * 32 + fkh * 16));
    v16h b2 = *(const v16h*)(bsrc + ((2 * 16 + fnl) * 32 + fkh * 16));
    v16h b3 = *(const v16h*)(bsrc + ((3 * 16 + fnl) * 32 + fkh * 16));

    c0 = __builtin_amdgcn_wmma_f32_16x16x32_f16(false, a, false, b0,
                                                (short)0, c0, false, false);
    c1 = __builtin_amdgcn_wmma_f32_16x16x32_f16(false, a, false, b1,
                                                (short)0, c1, false, false);
    c2 = __builtin_amdgcn_wmma_f32_16x16x32_f16(false, a, false, b2,
                                                (short)0, c2, false, false);
    c3 = __builtin_amdgcn_wmma_f32_16x16x32_f16(false, a, false, b3,
                                                (short)0, c3, false, false);

    asm volatile("s_wait_asynccnt 0x0" ::: "memory");
    __syncthreads();
  }

  // epilogue: D layout rows (lane>=16?8:0)+r, col lane&15; fuse bias
  const int rbase = (lane >> 4) * 8;
  const int col = lane & 15;
#pragma unroll
  for (int r = 0; r < 8; ++r) {
    const int co = mt * 128 + wave * 16 + rbase + r;
    const float bias = act_b[co];
    const size_t orow = ((size_t)(b * COUT + co) * OH + oh) * OW;
    Cv[orow + col]      = c0[r] + bias;
    Cv[orow + 16 + col] = c1[r] + bias;
    Cv[orow + 32 + col] = c2[r] + bias;
    if (col < 14) Cv[orow + 48 + col] = c3[r] + bias;  // ow 62/63 dropped
  }
}

// ---------------------------------------------------------------------------
// Kernel 5: fused upfirdn pipeline per (b,c) channel, entirely in LDS
//   62x62 -> up-h 62x128 -> up-v+lrelu 128x128 -> down-h 128x64 -> down-v
// up:   y[n] = sum_j fu[j] * xu[n+3-j]   (xu = 2x zero-stuffed, fu = filt*2)
// down: y[n] = sum_j fd[j] * x[2n+6-j]
// ---------------------------------------------------------------------------
__global__ __launch_bounds__(256) void firpipe_kernel(
    const float* __restrict__ Cv, const float* __restrict__ up_f,
    const float* __restrict__ dn_f, float* __restrict__ out) {
  __shared__ float smem[32512];  // 130,048 B (< 320 KB/WG on CDNA5)
  float* sA  = smem;             // 8192: input (3844), later down-h (8192)
  float* sUH = smem + 8192;      // 7936: up-h
  float* sUV = smem + 16128;     // 16384: up-v (+lrelu)
  const int bc = blockIdx.x;
  const int tid = threadIdx.x;

  float fu[12], fd[12];
#pragma unroll
  for (int j = 0; j < 12; ++j) { fu[j] = up_f[j] * 2.f; fd[j] = dn_f[j]; }

  const float* cin = Cv + (size_t)bc * (OH * OW);
  for (int i = tid; i < OH * OW; i += 256) sA[i] = cin[i];
  __syncthreads();

  for (int idx = tid; idx < 62 * 128; idx += 256) {  // up horizontal
    const int y = idx >> 7, ox = idx & 127;
    float acc = 0.f;
#pragma unroll
    for (int j = 0; j < 12; ++j) {
      const int mm = ox + 3 - j;
      if (!(mm & 1)) {
        const int xi = mm >> 1;
        if (xi >= 0 && xi < 62) acc += fu[j] * sA[y * 62 + xi];
      }
    }
    sUH[idx] = acc;
  }
  __syncthreads();

  for (int idx = tid; idx < 128 * 128; idx += 256) {  // up vertical + lrelu
    const int oy = idx >> 7, xx = idx & 127;
    float acc = 0.f;
#pragma unroll
    for (int j = 0; j < 12; ++j) {
      const int mm = oy + 3 - j;
      if (!(mm & 1)) {
        const int yi = mm >> 1;
        if (yi >= 0 && yi < 62) acc += fu[j] * sUH[yi * 128 + xx];
      }
    }
    sUV[idx] = (acc >= 0.f ? acc : 0.2f * acc) * 1.4142135623730951f;
  }
  __syncthreads();

  for (int idx = tid; idx < 128 * 64; idx += 256) {  // down horizontal
    const int y = idx >> 6, ox = idx & 63;
    float acc = 0.f;
#pragma unroll
    for (int j = 0; j < 12; ++j) {
      const int xi = 2 * ox + 6 - j;
      if (xi >= 0 && xi < 128) acc += fd[j] * sUV[y * 128 + xi];
    }
    sA[idx] = acc;
  }
  __syncthreads();

  float* op = out + (size_t)bc * 4096;
  for (int idx = tid; idx < 4096; idx += 256) {  // down vertical -> global
    const int oy = idx >> 6, xx = idx & 63;
    float acc = 0.f;
#pragma unroll
    for (int j = 0; j < 12; ++j) {
      const int yi = 2 * oy + 6 - j;
      if (yi >= 0 && yi < 128) acc += fd[j] * sA[yi * 64 + xx];
    }
    op[idx] = acc;
  }
}

// ---------------------------------------------------------------------------
extern "C" void kernel_launch(void* const* d_in, const int* in_sizes, int n_in,
                              void* d_out, int out_size, void* d_ws,
                              size_t ws_size, hipStream_t stream) {
  (void)in_sizes; (void)n_in; (void)out_size; (void)ws_size;
  const float* x      = (const float*)d_in[0];
  const float* style  = (const float*)d_in[1];
  const float* mod_w  = (const float*)d_in[2];
  const float* mod_b  = (const float*)d_in[3];
  const float* conv_w = (const float*)d_in[4];
  const float* act_b  = (const float*)d_in[5];
  const float* up_f   = (const float*)d_in[6];
  const float* dn_f   = (const float*)d_in[7];
  float* out = (float*)d_out;

  char* ws = (char*)d_ws;
  const size_t oW = 16384;                       // S: 16 KB
  const size_t oX = oW + (size_t)37748736;       // Wsw: 37.75 MB
  const size_t oC = oX + (size_t)33554432 + 4096;// Xh: 32 MB (+4 KB OOB pad)
  float*    S   = (float*)ws;
  _Float16* Wsw = (_Float16*)(ws + oW);
  _Float16* Xh  = (_Float16*)(ws + oX);
  float*    Cv  = (float*)(ws + oC);             // conv out: 63 MB

  style_kernel<<<B_, 256, 0, stream>>>(style, mod_w, mod_b, S);
  wpack_kernel<<<B_ * COUT, 256, 0, stream>>>(conv_w, S, Wsw);
  xcvt_kernel<<<B_ * H_, 256, 0, stream>>>(x, Xh);
  dim3 g3(OH, COUT / 128, B_);
  conv_wmma_kernel<<<g3, 256, 0, stream>>>(Xh, Wsw, act_b, Cv);
  firpipe_kernel<<<B_ * COUT, 256, 0, stream>>>(Cv, up_f, dn_f, out);
}
